// EGNN_7808250544487
// MI455X (gfx1250) — matrix operand order repaired and checked by reading, not compiled, not run on hardware
//
#include <hip/hip_runtime.h>
#include <hip/hip_bf16.h>
#include <math.h>

#define N_NODES 50000
#define N_EDGES 800000
#define N_GRAPH 1000
#define FDIM 16
#define HDIM 128
#define LAYERS 4
#define MSG_LDS_STRIDE 136  // 128 + 8 halfs padding -> 272B row stride, avoids LDS bank conflicts

typedef __attribute__((ext_vector_type(16))) _Float16 v16h;
typedef __attribute__((ext_vector_type(8)))  _Float16 v8h;
typedef __attribute__((ext_vector_type(8)))  float    v8f;

#define WMMA_F16(a, b, c) \
  __builtin_amdgcn_wmma_f32_16x16x32_f16(false, (a), false, (b), (short)0, (c), false, false)

__device__ __forceinline__ void atomAddF(float* p, float v) {
  __hip_atomic_fetch_add(p, v, __ATOMIC_RELAXED, __HIP_MEMORY_SCOPE_AGENT);
}

// A-fragment (16x32 f16) loader. `row` points to this lane's matrix row (M = lane&15).
// off = kslab*32 + (lane>>4)*8.  Elements 0..7 <- K=off..off+7, 8..15 <- K=off+16..off+23,
// matching the CDNA5 16-bit A-matrix VGPR layout.
__device__ __forceinline__ v16h load_afrag(const _Float16* row, int off) {
  v8h c0 = *(const v8h*)(row + off);
  v8h c1 = *(const v8h*)(row + off + 16);
  return __builtin_shufflevector(c0, c1, 0, 1, 2, 3, 4, 5, 6, 7, 8, 9, 10, 11, 12, 13, 14, 15);
}

// B-fragment from pre-packed buffer: 32 contiguous bytes per lane, coalesced.
__device__ __forceinline__ v16h load_bfrag(const _Float16* pk, int frag, int lane) {
  return *(const v16h*)(pk + (((size_t)frag * 32 + lane) << 4));
}

// ---------------------------------------------------------------------------
// Pre-pack a [K x 128] f32 weight matrix (layered) into per-lane WMMA
// B-fragment order as f16.  frag = t*(K/32) + ks ; element (lane,i) holds
// W[ks*32 + (lane>>4)*8 + (i&7) + (i>>3)*16][t*16 + (lane&15)].
__global__ void pack_bfrags_kernel(const float* __restrict__ W, _Float16* __restrict__ dst,
                                   int K, long srcStride, long dstStride) {
  int o = blockIdx.x * 256 + threadIdx.x;
  int total = K * HDIM;
  if (o >= total) return;
  long l = blockIdx.y;
  int i    = o & 15;
  int lane = (o >> 4) & 31;
  int frag = o >> 9;
  int nks  = K >> 5;
  int ks   = frag % nks;
  int t    = frag / nks;
  int kk   = ks * 32 + ((lane >> 4) << 3) + (i & 7) + ((i >> 3) << 4);
  int nn   = t * 16 + (lane & 15);
  dst[l * dstStride + o] = (_Float16)W[l * srcStride + (long)kk * HDIM + nn];
}

__global__ void zero_kernel(float* __restrict__ p, size_t n) {
  size_t i = (size_t)blockIdx.x * blockDim.x + threadIdx.x;
  size_t stride = (size_t)gridDim.x * blockDim.x;
  for (; i < n; i += stride) p[i] = 0.0f;
}

// ---------------------------------------------------------------------------
// edge_attr = ||pos[dst]-pos[src]||  and in-degree of dst (for mean aggregation)
__global__ void edge_prep_kernel(const float* __restrict__ pos, const int* __restrict__ esrc,
                                 const int* __restrict__ edst, float* __restrict__ eattr,
                                 float* __restrict__ deg) {
  int e = blockIdx.x * 256 + threadIdx.x;
  if (e >= N_EDGES) return;
  int s = esrc[e], d = edst[e];
  float dx = pos[d * 3 + 0] - pos[s * 3 + 0];
  float dy = pos[d * 3 + 1] - pos[s * 3 + 1];
  float dz = pos[d * 3 + 2] - pos[s * 3 + 2];
  eattr[e] = sqrtf(dx * dx + dy * dy + dz * dz);
  atomAddF(&deg[d], 1.0f);
}

// ---------------------------------------------------------------------------
// Embedder: h = relu(x@W1+b1)@W2+b2   (K=16 -> VALU is fine, <2 GFLOP)
__global__ __launch_bounds__(HDIM) void embed_kernel(
    const float* __restrict__ x, const float* __restrict__ W1, const float* __restrict__ b1,
    const float* __restrict__ W2, const float* __restrict__ b2,
    float* __restrict__ hf32, _Float16* __restrict__ hf16) {
  int n = blockIdx.x, j = threadIdx.x;
  __shared__ float xr[FDIM];
  __shared__ float t1[HDIM];
  if (j < FDIM) xr[j] = x[(long)n * FDIM + j];
  __syncthreads();
  float s = b1[j];
#pragma unroll
  for (int k = 0; k < FDIM; ++k) s += xr[k] * W1[k * HDIM + j];
  t1[j] = fmaxf(s, 0.0f);
  __syncthreads();
  float s2 = b2[j];
  for (int k = 0; k < HDIM; ++k) s2 += t1[k] * W2[k * HDIM + j];
  hf32[(long)n * HDIM + j] = s2;
  hf16[(long)n * HDIM + j] = (_Float16)s2;
}

// ---------------------------------------------------------------------------
// Message kernel: one wave per 32-edge tile (2 M-tiles of 16 sharing every
// B-fragment -> halves weight refetch traffic, doubles WMMA per loaded B).
// Whole 2-layer MLP + sigmoid gate + atomic scatter.
__global__ __launch_bounds__(128) void msg_kernel(
    const _Float16* __restrict__ hf16, const int* __restrict__ esrc,
    const int* __restrict__ edst, const float* __restrict__ eattr,
    const _Float16* __restrict__ pk1, const _Float16* __restrict__ pk2,
    const float* __restrict__ w256, const float* __restrict__ b1,
    const float* __restrict__ b2, const float* __restrict__ ew,
    const float* __restrict__ ebp, float* __restrict__ agg, int ntiles) {
  __shared__ __align__(32) _Float16 lm[4][32 * MSG_LDS_STRIDE];
  int lane = threadIdx.x & 31;
  int wv = threadIdx.x >> 5;
  int l15 = lane & 15, hlf = lane >> 4;
  int tile = blockIdx.x * 4 + wv;     // wave-uniform; EXEC stays all-ones for WMMA
  if (tile >= ntiles) return;
  int ebase = tile * 32;
  const _Float16* rowd0 = hf16 + (size_t)edst[ebase + l15] * HDIM;
  const _Float16* rowd1 = hf16 + (size_t)edst[ebase + 16 + l15] * HDIM;
  const _Float16* rows0 = hf16 + (size_t)esrc[ebase + l15] * HDIM;
  const _Float16* rows1 = hf16 + (size_t)esrc[ebase + 16 + l15] * HDIM;
  int aoff = hlf * 8;

  // acc = b1 (broadcast over rows) + edge_attr rank-1 term (W1 row 256)
  v8f acc[2][8];
  float ea[2][8];
#pragma unroll
  for (int mt = 0; mt < 2; ++mt)
#pragma unroll
    for (int r = 0; r < 8; ++r) ea[mt][r] = eattr[ebase + mt * 16 + hlf * 8 + r];
#pragma unroll
  for (int t = 0; t < 8; ++t) {
    float bv = b1[t * 16 + l15];
    float w = w256[t * 16 + l15];
#pragma unroll
    for (int mt = 0; mt < 2; ++mt)
#pragma unroll
      for (int r = 0; r < 8; ++r) acc[mt][t][r] = bv + ea[mt][r] * w;
  }
  // m1 = inp @ W1 : K 0..127 <- h[dst], K 128..255 <- h[src]
#pragma unroll
  for (int ks = 0; ks < 4; ++ks) {
    v16h a0 = load_afrag(rowd0, ks * 32 + aoff);
    v16h a1 = load_afrag(rowd1, ks * 32 + aoff);
#pragma unroll
    for (int t = 0; t < 8; ++t) {
      v16h b = load_bfrag(pk1, t * 8 + ks, lane);
      acc[0][t] = WMMA_F16(a0, b, acc[0][t]);
      acc[1][t] = WMMA_F16(a1, b, acc[1][t]);
    }
  }
#pragma unroll
  for (int ks = 0; ks < 4; ++ks) {
    v16h a0 = load_afrag(rows0, ks * 32 + aoff);
    v16h a1 = load_afrag(rows1, ks * 32 + aoff);
#pragma unroll
    for (int t = 0; t < 8; ++t) {
      v16h b = load_bfrag(pk1, t * 8 + 4 + ks, lane);
      acc[0][t] = WMMA_F16(a0, b, acc[0][t]);
      acc[1][t] = WMMA_F16(a1, b, acc[1][t]);
    }
  }
  // ReLU -> wave-private LDS (row-major 32 x 128, padded stride) for C->A transpose
  _Float16* m1 = &lm[wv][0];
#pragma unroll
  for (int mt = 0; mt < 2; ++mt)
#pragma unroll
    for (int t = 0; t < 8; ++t)
#pragma unroll
      for (int r = 0; r < 8; ++r)
        m1[(mt * 16 + r + hlf * 8) * MSG_LDS_STRIDE + t * 16 + l15] =
            (_Float16)fmaxf(acc[mt][t][r], 0.0f);

  // m2 = relu(m1 @ W2 + b2)
  const _Float16* m1r0 = m1 + l15 * MSG_LDS_STRIDE;
  const _Float16* m1r1 = m1 + (16 + l15) * MSG_LDS_STRIDE;
#pragma unroll
  for (int t = 0; t < 8; ++t) {
    float bv = b2[t * 16 + l15];
#pragma unroll
    for (int mt = 0; mt < 2; ++mt)
#pragma unroll
      for (int r = 0; r < 8; ++r) acc[mt][t][r] = bv;
  }
#pragma unroll
  for (int ks = 0; ks < 4; ++ks) {
    v16h a0 = load_afrag(m1r0, ks * 32 + aoff);
    v16h a1 = load_afrag(m1r1, ks * 32 + aoff);
#pragma unroll
    for (int t = 0; t < 8; ++t) {
      v16h b = load_bfrag(pk2, t * 4 + ks, lane);
      acc[0][t] = WMMA_F16(a0, b, acc[0][t]);
      acc[1][t] = WMMA_F16(a1, b, acc[1][t]);
    }
  }
#pragma unroll
  for (int mt = 0; mt < 2; ++mt)
#pragma unroll
    for (int t = 0; t < 8; ++t)
#pragma unroll
      for (int r = 0; r < 8; ++r) acc[mt][t][r] = fmaxf(acc[mt][t][r], 0.0f);

  // edge gate = sigmoid(m2 @ edge_W + edge_b): per-row dot, half-wave shuffle reduce
  float p[2][8];
#pragma unroll
  for (int mt = 0; mt < 2; ++mt)
#pragma unroll
    for (int r = 0; r < 8; ++r) p[mt][r] = 0.0f;
#pragma unroll
  for (int t = 0; t < 8; ++t) {
    float w = ew[t * 16 + l15];
#pragma unroll
    for (int mt = 0; mt < 2; ++mt)
#pragma unroll
      for (int r = 0; r < 8; ++r) p[mt][r] += acc[mt][t][r] * w;
  }
  float eb = ebp[0];
#pragma unroll
  for (int mt = 0; mt < 2; ++mt)
#pragma unroll
    for (int r = 0; r < 8; ++r) {
      float v = p[mt][r];
      v += __shfl_xor(v, 1, 32);
      v += __shfl_xor(v, 2, 32);
      v += __shfl_xor(v, 4, 32);
      v += __shfl_xor(v, 8, 32);
      p[mt][r] = 1.0f / (1.0f + expf(-(v + eb)));
    }
  // scatter gated messages: agg[dst] += m2 * gate  (f32 L2 atomics)
#pragma unroll
  for (int mt = 0; mt < 2; ++mt)
#pragma unroll
    for (int r = 0; r < 8; ++r) {
      int node = edst[ebase + mt * 16 + hlf * 8 + r];
      float* rp = agg + (size_t)node * HDIM + l15;
      float g = p[mt][r];
#pragma unroll
      for (int t = 0; t < 8; ++t) atomAddF(rp + t * 16, acc[mt][t][r] * g);
    }
}

// agg -> f16 with mean normalization (deg clamped to 1)
__global__ void aggdiv_kernel(const float* __restrict__ agg, const float* __restrict__ deg,
                              _Float16* __restrict__ aggf16) {
  size_t i = (size_t)blockIdx.x * 256 + threadIdx.x;
  if (i >= (size_t)N_NODES * HDIM) return;
  size_t n = i >> 7;
  aggf16[i] = (_Float16)(agg[i] / fmaxf(deg[n], 1.0f));
}

// ---------------------------------------------------------------------------
// Node update: h += relu([h|agg] @ U1 + b1) @ U2 + b2 ; one wave per 16-node tile
__global__ __launch_bounds__(256) void upd_kernel(
    float* __restrict__ hf32, _Float16* __restrict__ hf16,
    const _Float16* __restrict__ aggf16, const _Float16* __restrict__ pk1,
    const _Float16* __restrict__ pk2, const float* __restrict__ b1,
    const float* __restrict__ b2, int ntiles) {
  __shared__ __align__(32) _Float16 lm[8][16 * MSG_LDS_STRIDE];
  int lane = threadIdx.x & 31;
  int wv = threadIdx.x >> 5;
  int l15 = lane & 15, hlf = lane >> 4;
  int tile = blockIdx.x * 8 + wv;
  if (tile >= ntiles) return;
  int base = tile * 16;
  const _Float16* rowh = hf16 + (size_t)(base + l15) * HDIM;
  const _Float16* rowa = aggf16 + (size_t)(base + l15) * HDIM;
  int aoff = hlf * 8;

  v8f acc[8];
#pragma unroll
  for (int t = 0; t < 8; ++t) {
    float bv = b1[t * 16 + l15];
#pragma unroll
    for (int r = 0; r < 8; ++r) acc[t][r] = bv;
  }
#pragma unroll
  for (int ks = 0; ks < 4; ++ks) {
    v16h a = load_afrag(rowh, ks * 32 + aoff);
#pragma unroll
    for (int t = 0; t < 8; ++t) acc[t] = WMMA_F16(a, load_bfrag(pk1, t * 8 + ks, lane), acc[t]);
  }
#pragma unroll
  for (int ks = 0; ks < 4; ++ks) {
    v16h a = load_afrag(rowa, ks * 32 + aoff);
#pragma unroll
    for (int t = 0; t < 8; ++t) acc[t] = WMMA_F16(a, load_bfrag(pk1, t * 8 + 4 + ks, lane), acc[t]);
  }
  _Float16* m1 = &lm[wv][0];
#pragma unroll
  for (int t = 0; t < 8; ++t)
#pragma unroll
    for (int r = 0; r < 8; ++r)
      m1[(r + hlf * 8) * MSG_LDS_STRIDE + t * 16 + l15] = (_Float16)fmaxf(acc[t][r], 0.0f);
  const _Float16* m1row = m1 + l15 * MSG_LDS_STRIDE;
#pragma unroll
  for (int t = 0; t < 8; ++t) {
    float bv = b2[t * 16 + l15];
#pragma unroll
    for (int r = 0; r < 8; ++r) acc[t][r] = bv;
  }
#pragma unroll
  for (int ks = 0; ks < 4; ++ks) {
    v16h a = load_afrag(m1row, ks * 32 + aoff);
#pragma unroll
    for (int t = 0; t < 8; ++t) acc[t] = WMMA_F16(a, load_bfrag(pk2, t * 4 + ks, lane), acc[t]);
  }
  // residual update in f32 master, refresh f16 mirror
#pragma unroll
  for (int r = 0; r < 8; ++r) {
    size_t row = (size_t)(base + hlf * 8 + r);
    float* hp = hf32 + row * HDIM + l15;
    _Float16* hq = hf16 + row * HDIM + l15;
#pragma unroll
    for (int t = 0; t < 8; ++t) {
      float nv = hp[t * 16] + acc[t][r];
      hp[t * 16] = nv;
      hq[t * 16] = (_Float16)nv;
    }
  }
}

// ---------------------------------------------------------------------------
// node_dec + global_add_pool: g[batch[n]] += relu(h@D1+b1)@D2+b2
__global__ __launch_bounds__(256) void dec_kernel(
    const _Float16* __restrict__ hf16, const int* __restrict__ batch,
    const _Float16* __restrict__ pk1, const _Float16* __restrict__ pk2,
    const float* __restrict__ b1, const float* __restrict__ b2,
    float* __restrict__ gbuf, int ntiles) {
  __shared__ __align__(32) _Float16 lm[8][16 * MSG_LDS_STRIDE];
  int lane = threadIdx.x & 31;
  int wv = threadIdx.x >> 5;
  int l15 = lane & 15, hlf = lane >> 4;
  int tile = blockIdx.x * 8 + wv;
  if (tile >= ntiles) return;
  int base = tile * 16;
  const _Float16* rowh = hf16 + (size_t)(base + l15) * HDIM;
  int aoff = hlf * 8;

  v8f acc[8];
#pragma unroll
  for (int t = 0; t < 8; ++t) {
    float bv = b1[t * 16 + l15];
#pragma unroll
    for (int r = 0; r < 8; ++r) acc[t][r] = bv;
  }
#pragma unroll
  for (int ks = 0; ks < 4; ++ks) {
    v16h a = load_afrag(rowh, ks * 32 + aoff);
#pragma unroll
    for (int t = 0; t < 8; ++t) acc[t] = WMMA_F16(a, load_bfrag(pk1, t * 4 + ks, lane), acc[t]);
  }
  _Float16* m1 = &lm[wv][0];
#pragma unroll
  for (int t = 0; t < 8; ++t)
#pragma unroll
    for (int r = 0; r < 8; ++r)
      m1[(r + hlf * 8) * MSG_LDS_STRIDE + t * 16 + l15] = (_Float16)fmaxf(acc[t][r], 0.0f);
  const _Float16* m1row = m1 + l15 * MSG_LDS_STRIDE;
#pragma unroll
  for (int t = 0; t < 8; ++t) {
    float bv = b2[t * 16 + l15];
#pragma unroll
    for (int r = 0; r < 8; ++r) acc[t][r] = bv;
  }
#pragma unroll
  for (int ks = 0; ks < 4; ++ks) {
    v16h a = load_afrag(m1row, ks * 32 + aoff);
#pragma unroll
    for (int t = 0; t < 8; ++t) acc[t] = WMMA_F16(a, load_bfrag(pk2, t * 4 + ks, lane), acc[t]);
  }
#pragma unroll
  for (int r = 0; r < 8; ++r) {
    int gi = batch[base + hlf * 8 + r];
    float* gp = gbuf + (size_t)gi * HDIM + l15;
#pragma unroll
    for (int t = 0; t < 8; ++t) atomAddF(gp + t * 16, acc[t][r]);
  }
}

// ---------------------------------------------------------------------------
// Head: out = relu(g@W1+b1)@W2+b2 ; OUT=1, tiny -> VALU
__global__ __launch_bounds__(HDIM) void head_kernel(
    const float* __restrict__ g, const float* __restrict__ W1, const float* __restrict__ b1,
    const float* __restrict__ W2, const float* __restrict__ b2, float* __restrict__ out) {
  int gi = blockIdx.x, j = threadIdx.x;
  __shared__ float gv[HDIM];
  __shared__ float red[HDIM];
  gv[j] = g[(long)gi * HDIM + j];
  __syncthreads();
  float s = b1[j];
  for (int k = 0; k < HDIM; ++k) s += gv[k] * W1[k * HDIM + j];
  red[j] = fmaxf(s, 0.0f) * W2[j];
  __syncthreads();
  for (int st = 64; st > 0; st >>= 1) {
    if (j < st) red[j] += red[j + st];
    __syncthreads();
  }
  if (j == 0) out[gi] = red[0] + b2[0];
}

// ---------------------------------------------------------------------------
extern "C" void kernel_launch(void* const* d_in, const int* in_sizes, int n_in,
                              void* d_out, int out_size, void* d_ws, size_t ws_size,
                              hipStream_t stream) {
  (void)in_sizes; (void)n_in; (void)out_size; (void)ws_size;
  const float* x       = (const float*)d_in[0];
  const float* pos     = (const float*)d_in[1];
  const int*   eidx    = (const int*)d_in[2];
  const int*   batch   = (const int*)d_in[3];
  const float* emb_W1  = (const float*)d_in[4];
  const float* emb_b1  = (const float*)d_in[5];
  const float* emb_W2  = (const float*)d_in[6];
  const float* emb_b2  = (const float*)d_in[7];
  const float* msg_W1  = (const float*)d_in[8];
  const float* msg_b1  = (const float*)d_in[9];
  const float* msg_W2  = (const float*)d_in[10];
  const float* msg_b2  = (const float*)d_in[11];
  const float* edge_W  = (const float*)d_in[12];
  const float* edge_b  = (const float*)d_in[13];
  const float* upd_W1  = (const float*)d_in[14];
  const float* upd_b1  = (const float*)d_in[15];
  const float* upd_W2  = (const float*)d_in[16];
  const float* upd_b2  = (const float*)d_in[17];
  const float* dec_W1  = (const float*)d_in[18];
  const float* dec_b1  = (const float*)d_in[19];
  const float* dec_W2  = (const float*)d_in[20];
  const float* dec_b2  = (const float*)d_in[21];
  const float* head_W1 = (const float*)d_in[22];
  const float* head_b1 = (const float*)d_in[23];
  const float* head_W2 = (const float*)d_in[24];
  const float* head_b2 = (const float*)d_in[25];
  const int* esrc = eidx;
  const int* edst = eidx + N_EDGES;

  char* w = (char*)d_ws;
  size_t off = 0;
  auto alloc = [&](size_t bytes) -> void* {
    void* p = w + off;
    off = (off + bytes + 255) & ~(size_t)255;
    return p;
  };
  float*    hf32   = (float*)alloc((size_t)N_NODES * HDIM * 4);
  _Float16* hf16   = (_Float16*)alloc((size_t)N_NODES * HDIM * 2);
  float*    agg    = (float*)alloc((size_t)N_NODES * HDIM * 4);
  _Float16* aggf16 = (_Float16*)alloc((size_t)N_NODES * HDIM * 2);
  float*    deg    = (float*)alloc((size_t)N_NODES * 4);
  float*    eattr  = (float*)alloc((size_t)N_EDGES * 4);
  float*    gbuf   = (float*)alloc((size_t)N_GRAPH * HDIM * 4);
  _Float16* pkm1   = (_Float16*)alloc((size_t)LAYERS * 256 * HDIM * 2);
  _Float16* pkm2   = (_Float16*)alloc((size_t)LAYERS * 128 * HDIM * 2);
  _Float16* pku1   = (_Float16*)alloc((size_t)LAYERS * 256 * HDIM * 2);
  _Float16* pku2   = (_Float16*)alloc((size_t)LAYERS * 128 * HDIM * 2);
  _Float16* pkd1   = (_Float16*)alloc((size_t)128 * HDIM * 2);
  _Float16* pkd2   = (_Float16*)alloc((size_t)128 * HDIM * 2);

  // --- pack weights into WMMA B-fragment order (f16) ---
  dim3 pb256((256 * HDIM + 255) / 256, LAYERS);
  dim3 pb128((128 * HDIM + 255) / 256, LAYERS);
  dim3 pbd((128 * HDIM + 255) / 256, 1);
  pack_bfrags_kernel<<<pb256, 256, 0, stream>>>(msg_W1, pkm1, 256, 257 * HDIM, 256 * HDIM);
  pack_bfrags_kernel<<<pb128, 256, 0, stream>>>(msg_W2, pkm2, 128, 128 * HDIM, 128 * HDIM);
  pack_bfrags_kernel<<<pb256, 256, 0, stream>>>(upd_W1, pku1, 256, 256 * HDIM, 256 * HDIM);
  pack_bfrags_kernel<<<pb128, 256, 0, stream>>>(upd_W2, pku2, 128, 128 * HDIM, 128 * HDIM);
  pack_bfrags_kernel<<<pbd, 256, 0, stream>>>(dec_W1, pkd1, 128, 0, 0);
  pack_bfrags_kernel<<<pbd, 256, 0, stream>>>(dec_W2, pkd2, 128, 0, 0);

  // --- graph-structure prep + embedder ---
  zero_kernel<<<256, 256, 0, stream>>>(deg, (size_t)N_NODES);
  edge_prep_kernel<<<(N_EDGES + 255) / 256, 256, 0, stream>>>(pos, esrc, edst, eattr, deg);
  embed_kernel<<<N_NODES, HDIM, 0, stream>>>(x, emb_W1, emb_b1, emb_W2, emb_b2, hf32, hf16);

  const int etiles = N_EDGES / 32;  // 25000 (32 edges per wave)
  const int ntile  = N_NODES / 16;  // 3125
  for (int l = 0; l < LAYERS; ++l) {
    zero_kernel<<<1024, 256, 0, stream>>>(agg, (size_t)N_NODES * HDIM);
    msg_kernel<<<(etiles + 3) / 4, 128, 0, stream>>>(
        hf16, esrc, edst, eattr,
        pkm1 + (size_t)l * 256 * HDIM, pkm2 + (size_t)l * 128 * HDIM,
        msg_W1 + ((size_t)l * 257 + 256) * HDIM, msg_b1 + l * HDIM, msg_b2 + l * HDIM,
        edge_W + l * HDIM, edge_b + l, agg, etiles);
    aggdiv_kernel<<<(int)(((size_t)N_NODES * HDIM + 255) / 256), 256, 0, stream>>>(agg, deg, aggf16);
    upd_kernel<<<(ntile + 7) / 8, 256, 0, stream>>>(
        hf32, hf16, aggf16,
        pku1 + (size_t)l * 256 * HDIM, pku2 + (size_t)l * 128 * HDIM,
        upd_b1 + l * HDIM, upd_b2 + l * HDIM, ntile);
  }
  zero_kernel<<<64, 256, 0, stream>>>(gbuf, (size_t)N_GRAPH * HDIM);
  dec_kernel<<<(ntile + 7) / 8, 256, 0, stream>>>(hf16, batch, pkd1, pkd2, dec_b1, dec_b2,
                                                  gbuf, ntile);
  head_kernel<<<N_GRAPH, HDIM, 0, stream>>>(gbuf, head_W1, head_b1, head_W2, head_b2,
                                            (float*)d_out);
}
